// LSTMN_9070970929482
// MI455X (gfx1250) — matrix operand order repaired and checked by reading, not compiled, not run on hardware
//
#include <hip/hip_runtime.h>
#include <hip/hip_bf16.h>

typedef __attribute__((ext_vector_type(16))) __bf16 v16bf;
typedef __attribute__((ext_vector_type(8)))  float  v8f;

#define BATCH 128
#define DIM   300
#define TLEN  1024

__device__ __forceinline__ v8f wmma_bf16(v16bf a, v16bf b, v8f c) {
    // emits v_wmma_f32_16x16x32_bf16
    return __builtin_amdgcn_wmma_f32_16x16x32_bf16(
        /*neg_a=*/false, a, /*neg_b=*/false, b,
        /*c_mod=*/(short)0, c, /*reuse_a=*/false, /*reuse_b=*/false);
}

// ---------------------------------------------------------------------------
// 1) xw[b][j] = dot(x[b], W_x[j])   (tiny; one wave per batch row)
// ---------------------------------------------------------------------------
__global__ void xw_kernel(const float* __restrict__ x, const float* __restrict__ Wx,
                          float* __restrict__ xw) {
    int b = blockIdx.x, lane = threadIdx.x;
    for (int j = 0; j < 10; ++j) {
        float a = 0.f;
        for (int d = lane; d < DIM; d += 32) a += x[b * DIM + d] * Wx[j * DIM + d];
        for (int o = 16; o; o >>= 1) a += __shfl_down(a, o, 32);
        if (lane == 0) xw[b * 10 + j] = a;
    }
}

// ---------------------------------------------------------------------------
// 2) scores[b][t] = v . tanh(W_h h[b,t] + W_x x[b] + W_ht pht[b,t])
//    One wave per (b,t); coalesced streaming over d; HBM-bound by design.
// ---------------------------------------------------------------------------
__global__ void scores_kernel(const float* __restrict__ hs, const float* __restrict__ pht,
                              const float* __restrict__ Wh, const float* __restrict__ Wht,
                              const float* __restrict__ v,  const float* __restrict__ xw,
                              float* __restrict__ sc) {
    __shared__ float lwh[10 * DIM];
    __shared__ float lwt[10 * DIM];
    __shared__ float lv[16], lxw[16];
    int tid = threadIdx.x, b = blockIdx.y;
    for (int i = tid; i < 10 * DIM; i += 256) { lwh[i] = Wh[i]; lwt[i] = Wht[i]; }
    if (tid < 10) { lv[tid] = v[tid]; lxw[tid] = xw[b * 10 + tid]; }
    __syncthreads();

    int lane = tid & 31, wv = tid >> 5;
    int t = blockIdx.x * 8 + wv;
    const float* hrow = hs  + ((size_t)b * TLEN + t) * DIM;
    const float* prow = pht + ((size_t)b * TLEN + t) * DIM;

    float aj[10];
#pragma unroll
    for (int j = 0; j < 10; ++j) aj[j] = 0.f;
    for (int d = lane; d < DIM; d += 32) {
        float hd = hrow[d], pd = prow[d];
#pragma unroll
        for (int j = 0; j < 10; ++j) aj[j] += lwh[j * DIM + d] * hd + lwt[j * DIM + d] * pd;
    }
    float s = 0.f;
#pragma unroll
    for (int j = 0; j < 10; ++j) {
        float a = aj[j];
        for (int o = 16; o; o >>= 1) a += __shfl_down(a, o, 32);
        a = __shfl(a, 0, 32);
        s += lv[j] * tanhf(a + lxw[j]);
    }
    if (lane == 0) sc[(size_t)b * TLEN + t] = s;
}

// ---------------------------------------------------------------------------
// 3) softmax over T per batch row
// ---------------------------------------------------------------------------
__global__ void softmax_kernel(float* __restrict__ s) {
    __shared__ float red[256];
    int b = blockIdx.x, tid = threadIdx.x;
    float v0[4];
    float mx = -3.0e38f;
#pragma unroll
    for (int i = 0; i < 4; ++i) { v0[i] = s[(size_t)b * TLEN + tid + i * 256]; mx = fmaxf(mx, v0[i]); }
    red[tid] = mx; __syncthreads();
    for (int o = 128; o; o >>= 1) { if (tid < o) red[tid] = fmaxf(red[tid], red[tid + o]); __syncthreads(); }
    mx = red[0]; __syncthreads();
    float sum = 0.f;
#pragma unroll
    for (int i = 0; i < 4; ++i) { v0[i] = expf(v0[i] - mx); sum += v0[i]; }
    red[tid] = sum; __syncthreads();
    for (int o = 128; o; o >>= 1) { if (tid < o) red[tid] += red[tid + o]; __syncthreads(); }
    float inv = 1.f / red[0];
#pragma unroll
    for (int i = 0; i < 4; ++i) s[(size_t)b * TLEN + tid + i * 256] = v0[i] * inv;
}

// ---------------------------------------------------------------------------
// 4) ht = sum_t s*h, ct = sum_t s*c; write hx = [ht | x]
// ---------------------------------------------------------------------------
__global__ void wsum_kernel(const float* __restrict__ sc, const float* __restrict__ hs,
                            const float* __restrict__ pc, const float* __restrict__ x,
                            float* __restrict__ hx, float* __restrict__ ct) {
    __shared__ float ls[TLEN];
    int b = blockIdx.x, tid = threadIdx.x;
    for (int i = tid; i < TLEN; i += 320) ls[i] = sc[(size_t)b * TLEN + i];
    __syncthreads();
    if (tid < DIM) {
        float ah = 0.f, ac = 0.f;
        const float* hb = hs + (size_t)b * TLEN * DIM + tid;
        const float* cb = pc + (size_t)b * TLEN * DIM + tid;
        for (int t = 0; t < TLEN; ++t) {
            float s = ls[t];
            ah += s * hb[(size_t)t * DIM];
            ac += s * cb[(size_t)t * DIM];
        }
        hx[b * 2 * DIM + tid]       = ah;
        hx[b * 2 * DIM + DIM + tid] = x[b * DIM + tid];
        ct[b * DIM + tid]           = ac;
    }
}

// ---------------------------------------------------------------------------
// 5) bf16-WMMA GEMM:  out[M,N] = A[M,K] @ W[N,K]^T + bias
//    128 threads = 4 waves; each wave owns a 16-row tile; K chunked through LDS.
//    Weight tile stored fragment-major: lw[((k>>4)*16 + n)*16 + (k&15)] so each
//    lane's 16 B elements are 32 contiguous, 32B-aligned bytes -> ds_load_b128.
// ---------------------------------------------------------------------------
#define GKC 128
__global__ void gemm_bf16_wmma(const float* __restrict__ A, const float* __restrict__ Wt,
                               const float* __restrict__ bias,
                               float* __restrict__ outF, __bf16* __restrict__ outB,
                               int M, int N, int K) {
    __shared__ __bf16 lw[GKC * 16];
    __shared__ __bf16 la[64 * GKC];
    int tid = threadIdx.x, lane = tid & 31, wv = tid >> 5;
    int half = lane >> 4, l15 = lane & 15;
    int mbase = blockIdx.x * 64, nbase = blockIdx.y * 16;
    v8f acc = {0.f, 0.f, 0.f, 0.f, 0.f, 0.f, 0.f, 0.f};
    int nch = (K + GKC - 1) / GKC;
    for (int c = 0; c < nch; ++c) {
        int kb = c * GKC;
        for (int idx = tid; idx < 64 * GKC; idx += 128) {
            int m = idx / GKC, k = idx % GKC;
            int gm = mbase + m, gk = kb + k;
            la[idx] = (__bf16)((gm < M && gk < K) ? A[(size_t)gm * K + gk] : 0.f);
        }
        for (int idx = tid; idx < 16 * GKC; idx += 128) {
            int n = idx / GKC, k = idx % GKC;
            int gn = nbase + n, gk = kb + k;
            float vv = (gn < N && gk < K) ? Wt[(size_t)gn * K + gk] : 0.f;
            lw[((k >> 4) * 16 + n) * 16 + (k & 15)] = (__bf16)vv;
        }
        __syncthreads();
        for (int ks = 0; ks < GKC / 32; ++ks) {
            v16bf af, bf;
            int abase = (wv * 16 + l15) * GKC + ks * 32 + half * 16;
            int bbase = ((ks * 2 + half) * 16 + l15) * 16;
#pragma unroll
            for (int e = 0; e < 8; ++e) {
                af[e]     = la[abase + e];
                af[8 + e] = la[abase + 8 + e];
            }
#pragma unroll
            for (int e = 0; e < 16; ++e) bf[e] = lw[bbase + e];
            acc = wmma_bf16(af, bf, acc);
        }
        __syncthreads();
    }
    int col = nbase + l15;
#pragma unroll
    for (int r = 0; r < 8; ++r) {
        int row = mbase + wv * 16 + r + 8 * half;
        if (row < M && col < N) {
            float vv = acc[r] + (bias ? bias[col] : 0.f);
            if (outF) outF[(size_t)row * N + col] = vv;
            else      outB[(size_t)row * N + col] = (__bf16)vv;
        }
    }
}

// ---------------------------------------------------------------------------
// 6) Implicit-GEMM 3x3 SAME conv, bf16-WMMA, fused bias+ReLU+BN epilogue.
//    M = 16-pixel tiles (lanes 0-15 gather 16 consecutive pixels -> coalesced),
//    N = 16 output channels, K = ci*9 staged in LDS fragment-major (as above).
//    concat2: channel 0 from inA, channel 1 from inB (for the pc conv).
// ---------------------------------------------------------------------------
__global__ void conv_bf16_wmma(const __bf16* __restrict__ inA, const __bf16* __restrict__ inB,
                               int concat2, int ci, int co,
                               const float* __restrict__ w, const float* __restrict__ bias,
                               const float* __restrict__ g, const float* __restrict__ beta,
                               int do_relu, float* __restrict__ outF, __bf16* __restrict__ outB) {
    __shared__ __bf16 lw[576 * 16];
    __shared__ int lkc[576], ldy[576], ldx[576];
    int K = ci * 9, K32 = (K + 31) & ~31;
    int tid = threadIdx.x;
    for (int k = tid; k < K32; k += 128) {
        if (k < K) { int c = k / 9, rr = k - c * 9; lkc[k] = c; ldy[k] = rr / 3 - 1; ldx[k] = rr % 3 - 1; }
        else       { lkc[k] = -1; ldy[k] = 0; ldx[k] = 0; }
    }
    int nbase = blockIdx.y * 16;
    for (int idx = tid; idx < 16 * K32; idx += 128) {
        int n = idx / K32, k = idx % K32;
        int gn = nbase + n;
        float vv = (gn < co && k < K) ? w[(size_t)gn * K + k] : 0.f;
        lw[((k >> 4) * 16 + n) * 16 + (k & 15)] = (__bf16)vv;
    }
    __syncthreads();

    int lane = tid & 31, wv = tid >> 5;
    int half = lane >> 4, l15 = lane & 15;
    int b = blockIdx.z;
    int ptile = blockIdx.x * 4 + wv;
    int p = ptile * 16 + l15;
    int py = p / 40, px = p - py * 40;
    v8f acc = {0.f, 0.f, 0.f, 0.f, 0.f, 0.f, 0.f, 0.f};
    for (int ks = 0; ks < K32 / 32; ++ks) {
        v16bf af, bf;
#pragma unroll
        for (int e = 0; e < 16; ++e) {
            int ka = ks * 32 + half * 16 + e;
            int c = lkc[ka];
            float av = 0.f;
            if (c >= 0) {
                int iy = py + ldy[ka], ix = px + ldx[ka];
                if ((unsigned)iy < 40u && (unsigned)ix < 40u) {
                    int pp = iy * 40 + ix;
                    av = concat2
                       ? (float)(c ? inB[(size_t)b * 1600 + pp] : inA[(size_t)b * 1600 + pp])
                       : (float)inA[((size_t)b * ci + c) * 1600 + pp];
                }
            }
            af[e] = (__bf16)av;
        }
        int bbase = ((ks * 2 + half) * 16 + l15) * 16;
#pragma unroll
        for (int e = 0; e < 16; ++e) bf[e] = lw[bbase + e];
        acc = wmma_bf16(af, bf, acc);
    }
    int ch = nbase + l15;
    if (ch < co) {
        float sc = g[ch] * rsqrtf(1.0f + 1e-5f);
        float bt = beta[ch], bs = bias[ch];
#pragma unroll
        for (int r = 0; r < 8; ++r) {
            int pr = ptile * 16 + r + 8 * half;
            float vv = acc[r] + bs;
            if (do_relu) vv = fmaxf(vv, 0.f);
            vv = vv * sc + bt;
            if (outF) outF[((size_t)b * co + ch) * 1600 + pr] = vv;
            else      outB[((size_t)b * co + ch) * 1600 + pr] = (__bf16)vv;
        }
    }
}

// ---------------------------------------------------------------------------
// 7) LSTM gating -> h
// ---------------------------------------------------------------------------
__global__ void gate_kernel(const float* __restrict__ whx, const float* __restrict__ ct,
                            float* __restrict__ out) {
    int b = blockIdx.x, d = threadIdx.x;
    if (d < DIM) {
        const float* r = whx + (size_t)b * 4 * DIM;
        float f = 1.f / (1.f + expf(-r[d]));
        float o = 1.f / (1.f + expf(-r[DIM + d]));
        float i = 1.f / (1.f + expf(-r[2 * DIM + d]));
        float chg = tanhf(r[3 * DIM + d]);
        float c = f * ct[b * DIM + d] + i * chg;
        out[b * DIM + d] = o * tanhf(c);
    }
}

// ---------------------------------------------------------------------------
extern "C" void kernel_launch(void* const* d_in, const int* in_sizes, int n_in,
                              void* d_out, int out_size, void* d_ws, size_t ws_size,
                              hipStream_t stream) {
    (void)in_sizes; (void)n_in; (void)out_size; (void)ws_size;
    const float* x    = (const float*)d_in[0];
    const float* hs   = (const float*)d_in[1];
    const float* pht  = (const float*)d_in[2];
    const float* pc   = (const float*)d_in[3];
    const float* W    = (const float*)d_in[4];
    const float* bvec = (const float*)d_in[5];
    const float* Wh   = (const float*)d_in[6];
    const float* Wx   = (const float*)d_in[7];
    const float* Wht  = (const float*)d_in[8];
    const float* vv   = (const float*)d_in[9];
    const float* rb_w = (const float*)d_in[10];
    const float* rb_b = (const float*)d_in[11];
    const float* rp_w = (const float*)d_in[12];
    const float* rp_b = (const float*)d_in[13];
    const float* cw[7], *cb[7], *cg[7], *cbe[7];
    for (int l = 0; l < 6; ++l) {
        cw[l]  = (const float*)d_in[14 + 4 * l];
        cb[l]  = (const float*)d_in[15 + 4 * l];
        cg[l]  = (const float*)d_in[16 + 4 * l];
        cbe[l] = (const float*)d_in[17 + 4 * l];
    }
    cw[6] = (const float*)d_in[38]; cb[6] = (const float*)d_in[39];
    cg[6] = (const float*)d_in[40]; cbe[6] = (const float*)d_in[41];

    char* wsb = (char*)d_ws;
    size_t off = 0;
    auto alloc = [&](size_t bytes) -> char* {
        char* p = wsb + off; off += (bytes + 255) & ~(size_t)255; return p;
    };
    float*  xw    = (float*) alloc((size_t)BATCH * 10 * 4);
    float*  sc    = (float*) alloc((size_t)BATCH * TLEN * 4);
    float*  hx    = (float*) alloc((size_t)BATCH * 2 * DIM * 4);
    float*  ct    = (float*) alloc((size_t)BATCH * DIM * 4);
    float*  whx1  = (float*) alloc((size_t)BATCH * 4 * DIM * 4);
    __bf16* pre   = (__bf16*)alloc((size_t)BATCH * 1600 * 2);
    __bf16* bufA  = (__bf16*)alloc((size_t)BATCH * 64 * 1600 * 2);
    __bf16* bufB  = (__bf16*)alloc((size_t)BATCH * 64 * 1600 * 2);
    __bf16* y6    = (__bf16*)alloc((size_t)BATCH * 1600 * 2);
    float*  ypc   = (float*) alloc((size_t)BATCH * 1600 * 4);
    float*  whx3  = (float*) alloc((size_t)BATCH * 4 * DIM * 4);

    // --- attention (HBM-bound, VALU streaming) ---
    xw_kernel<<<BATCH, 32, 0, stream>>>(x, Wx, xw);
    scores_kernel<<<dim3(TLEN / 8, BATCH), 256, 0, stream>>>(hs, pht, Wh, Wht, vv, xw, sc);
    softmax_kernel<<<BATCH, 256, 0, stream>>>(sc);
    wsum_kernel<<<BATCH, 320, 0, stream>>>(sc, hs, pc, x, hx, ct);

    // --- dense layers (bf16 WMMA) ---
    gemm_bf16_wmma<<<dim3(2, 75), 128, 0, stream>>>(hx,   W,    bvec, whx1,    nullptr, BATCH, 4 * DIM, 2 * DIM);
    gemm_bf16_wmma<<<dim3(2, 100), 128, 0, stream>>>(whx1, rb_w, rb_b, nullptr, pre,     BATCH, 1600,    4 * DIM);

    // --- conv stack (implicit GEMM, bf16 WMMA, fused bias/relu/bn) ---
    conv_bf16_wmma<<<dim3(25, 2, BATCH), 128, 0, stream>>>(pre,  nullptr, 0, 1,  32, cw[0], cb[0], cg[0], cbe[0], 1, nullptr, bufA);
    conv_bf16_wmma<<<dim3(25, 4, BATCH), 128, 0, stream>>>(bufA, nullptr, 0, 32, 64, cw[1], cb[1], cg[1], cbe[1], 1, nullptr, bufB);
    conv_bf16_wmma<<<dim3(25, 4, BATCH), 128, 0, stream>>>(bufB, nullptr, 0, 64, 64, cw[2], cb[2], cg[2], cbe[2], 1, nullptr, bufA);
    conv_bf16_wmma<<<dim3(25, 4, BATCH), 128, 0, stream>>>(bufA, nullptr, 0, 64, 64, cw[3], cb[3], cg[3], cbe[3], 1, nullptr, bufB);
    conv_bf16_wmma<<<dim3(25, 2, BATCH), 128, 0, stream>>>(bufB, nullptr, 0, 64, 32, cw[4], cb[4], cg[4], cbe[4], 1, nullptr, bufA);
    conv_bf16_wmma<<<dim3(25, 1, BATCH), 128, 0, stream>>>(bufA, nullptr, 0, 32, 1,  cw[5], cb[5], cg[5], cbe[5], 1, nullptr, y6);
    conv_bf16_wmma<<<dim3(25, 1, BATCH), 128, 0, stream>>>(pre,  y6,      1, 2,  1,  cw[6], cb[6], cg[6], cbe[6], 0, ypc,    nullptr);

    // --- reshape_pos GEMM + gating ---
    gemm_bf16_wmma<<<dim3(2, 75), 128, 0, stream>>>(ypc, rp_w, rp_b, whx3, nullptr, BATCH, 4 * DIM, 1600);
    gate_kernel<<<BATCH, 320, 0, stream>>>(whx3, ct, (float*)d_out);
}
